// DeltaSynapse_84550726189209
// MI455X (gfx1250) — matrix operand order repaired and checked by reading, not compiled, not run on hardware
//
#include <hip/hip_runtime.h>
#include <hip/hip_bf16.h>

typedef __attribute__((ext_vector_type(16))) _Float16 v16h;
typedef __attribute__((ext_vector_type(8)))  _Float16 v8h;
typedef __attribute__((ext_vector_type(8)))  float    v8f;

#define DD 8
#define BB 16
#define EE 2048
#define OO 2048
#define SEGS 16                       // split of E dimension for parallelism
#define E_PER_SEG (EE / SEGS)         // 128
#define CHUNK 32                      // K per WMMA
#define CHUNKS_PER_SEG (E_PER_SEG / CHUNK)  // 4
#define OTILES (OO / 16)              // 128
#define WAVES_PER_BLOCK 8

// ---------------------------------------------------------------------------
// prep: A_h[d,b,e] = half( Xd[d,b,e] * (1 + Wshort[d,b,e]) )
// ---------------------------------------------------------------------------
__global__ void prep_A_kernel(const float* __restrict__ Xd,
                              const float* __restrict__ Wshort,
                              _Float16* __restrict__ Ah, int n) {
    int i = blockIdx.x * blockDim.x + threadIdx.x;
    if (i < n) Ah[i] = (_Float16)(Xd[i] * (1.0f + Wshort[i]));
}

// ---------------------------------------------------------------------------
// main: per wave, one 16-col o-tile x one 128-row e-segment.
// acc(16x16 f32) += A(16x32 f16) x B(32x16 f16), looped over d and e-chunks.
// B fragment = delaymap ? sign(e)*relu(W[e,o]) : 0   (binary mask -> select, exact)
// ---------------------------------------------------------------------------
__global__ __launch_bounds__(256)
void synapse_wmma_kernel(const float* __restrict__ W,
                         const float* __restrict__ delaymap,
                         const int*   __restrict__ signs,
                         const _Float16* __restrict__ Ah,
                         float* __restrict__ part) {
    const int wid  = threadIdx.x >> 5;
    const int lane = threadIdx.x & 31;
    const int task = blockIdx.x * WAVES_PER_BLOCK + wid;   // 0..2047
    const int ot   = task >> 4;                            // 0..127 o-tile
    const int seg  = task & (SEGS - 1);                    // 0..15  e-segment
    const int o0    = ot << 4;
    const int ebase = seg * E_PER_SEG;
    const int hs = lane >> 4;     // half-wave select (0/1)
    const int n  = lane & 15;     // column within tile / row for A

    v8f acc = {};

    for (int ec = 0; ec < CHUNKS_PER_SEG; ++ec) {
        const int e0  = ebase + ec * CHUNK;
        const int klo = hs << 4;  // B-operand K offset per half-wave: 0 or 16

        // Signed f16 weight column for this e-chunk / o-column, kept in VGPRs
        // across the d-loop so W (+signs) is read exactly once per (e,o).
        v16h wsh;
#pragma unroll
        for (int i = 0; i < 16; ++i) {
            const int e = e0 + klo + i;
            const float w = W[(size_t)e * OO + o0 + n];
            const float s = (float)(2 * signs[e] - 1);
            wsh[i] = (w > 0.0f) ? (_Float16)(w * s) : (_Float16)0.0f;
        }

#pragma unroll
        for (int d = 0; d < DD; ++d) {
            // B fragment: binary delaymap gates the signed weight (select, no mul)
            const float* dmp =
                delaymap + ((size_t)(d * EE + e0 + klo)) * OO + o0 + n;
            v16h bmat;
#pragma unroll
            for (int i = 0; i < 16; ++i) {
                const float dm = dmp[(size_t)i * OO];
                bmat[i] = (dm > 0.0f) ? wsh[i] : (_Float16)0.0f;
            }

            // A fragment (contiguous k order, same as B): row = n,
            // lanes 0-15: K = {0..15};  lanes 16-31: K = {16..31}
            const _Float16* ap = Ah + ((size_t)(d * BB + n)) * EE + e0 + hs * 16;
            const v8h alo = *(const v8h*)(ap);
            const v8h ahi = *(const v8h*)(ap + 8);
            v16h amat;
#pragma unroll
            for (int i = 0; i < 8; ++i) { amat[i] = alo[i]; amat[8 + i] = ahi[i]; }

            acc = __builtin_amdgcn_wmma_f32_16x16x32_f16(
                /*neg_a=*/false, amat, /*neg_b=*/false, bmat,
                /*c_mod=*/(short)0, acc, /*reuse_a=*/false, /*reuse_b=*/false);
        }
    }

    // C/D layout: VGPR r -> row r (lanes 0-15) / row r+8 (lanes 16-31), col = n
#pragma unroll
    for (int r = 0; r < 8; ++r) {
        const int row = r + hs * 8;
        part[((size_t)seg * BB + row) * OO + o0 + n] = acc[r];
    }
}

// ---------------------------------------------------------------------------
// deterministic reduction over e-segments: out[b,o] = sum_seg part[seg,b,o]
// ---------------------------------------------------------------------------
__global__ void reduce_part_kernel(const float* __restrict__ part,
                                   float* __restrict__ out, int n) {
    int i = blockIdx.x * blockDim.x + threadIdx.x;
    if (i < n) {
        float s = 0.0f;
#pragma unroll
        for (int g = 0; g < SEGS; ++g) s += part[(size_t)g * n + i];
        out[i] = s;
    }
}

// ---------------------------------------------------------------------------
extern "C" void kernel_launch(void* const* d_in, const int* in_sizes, int n_in,
                              void* d_out, int out_size, void* d_ws, size_t ws_size,
                              hipStream_t stream) {
    const float* W        = (const float*)d_in[0];   // (E,O)
    const float* Xd       = (const float*)d_in[1];   // (D,B,E)
    const float* delaymap = (const float*)d_in[2];   // (D,E,O)
    const float* Wshort   = (const float*)d_in[3];   // (D,B,E)
    const int*   signs    = (const int*)d_in[4];     // (E,)
    float* out = (float*)d_out;                      // (B,O) f32

    // workspace layout
    _Float16* Ah  = (_Float16*)d_ws;                          // D*B*E halfs = 512 KB
    float*    prt = (float*)((char*)d_ws + (size_t)DD * BB * EE * sizeof(_Float16));
                                                              // SEGS*B*O f32 = 2 MB

    // 1) A_h = Xd * (1 + Wshort) in f16
    {
        const int n = DD * BB * EE;
        prep_A_kernel<<<(n + 255) / 256, 256, 0, stream>>>(Xd, Wshort, Ah, n);
    }

    // 2) WMMA contraction over (d, e) into per-segment partials
    {
        const int tasks  = OTILES * SEGS;                 // 2048 wave-tasks
        const int blocks = tasks / WAVES_PER_BLOCK;       // 256 blocks x 256 thr
        synapse_wmma_kernel<<<blocks, WAVES_PER_BLOCK * 32, 0, stream>>>(
            W, delaymap, signs, Ah, prt);
    }

    // 3) reduce segments -> output
    {
        const int n = BB * OO;                            // 32768
        reduce_part_kernel<<<(n + 255) / 256, 256, 0, stream>>>(prt, out, n);
    }
}
